// PUSCHNeuralDetector_69423851373132
// MI455X (gfx1250) — matrix operand order repaired and checked by reading, not compiled, not run on hardware
//
#include <hip/hip_runtime.h>
#include <hip/hip_bf16.h>

// ---------------------------------------------------------------------------
// CDNA5 / gfx1250 WMMA types
// ---------------------------------------------------------------------------
typedef __attribute__((ext_vector_type(16))) __bf16 v16bf;
typedef __attribute__((ext_vector_type(8)))  __bf16 v8bf;
typedef __attribute__((ext_vector_type(8)))  float  v8f;

// Problem constants (from the reference)
#define BB   2
#define TT   14
#define LL   1536      // spatial length (the "H" axis; W==1)
#define C0   297       // raw input channels
#define C0P  320       // padded to multiple of 32
#define F1   64        // FILTERS
#define N1   256       // 4*F1
#define F4   32        // OUT_CH
#define N4   128       // 4*F4

// ---------------------------------------------------------------------------
// Weight repack: f32 [3,3,Cin,Nout] -> bf16 Wt[tap][n][Kpad] (center column,
// kw==1, since W==1 with SAME padding zero-pads the horizontal taps away).
// K-contiguous per output channel so a lane's B-fragment is one 32B load.
// ---------------------------------------------------------------------------
__global__ void repack_w(const float* __restrict__ src, __bf16* __restrict__ dst,
                         int Cin, int Kpad, int Nout) {
    long long idx = (long long)blockIdx.x * blockDim.x + threadIdx.x;
    long long total = 3LL * Nout * Kpad;
    if (idx >= total) return;
    int k   = (int)(idx % Kpad);
    int n   = (int)((idx / Kpad) % Nout);
    int tap = (int)(idx / ((long long)Kpad * Nout));
    float v = (k < Cin) ? src[(((size_t)(tap * 3 + 1)) * Cin + k) * Nout + n] : 0.0f;
    dst[idx] = (__bf16)v;
}

// ---------------------------------------------------------------------------
// Input feature packing: fused transpose/concat of y_ri, h_ri, err_var, no
// into bf16 X0[(b*T+t)][l][C0P], zero-padded channels [297,320).
// ---------------------------------------------------------------------------
__global__ void pack_input(const float* __restrict__ y, const float* __restrict__ h,
                           const float* __restrict__ ev, const float* __restrict__ no,
                           __bf16* __restrict__ X) {
    long long idx = (long long)blockIdx.x * blockDim.x + threadIdx.x;
    long long total = (long long)BB * TT * LL * C0P;
    if (idx >= total) return;
    int c = (int)(idx % C0P);
    long long r = idx / C0P;
    int l = (int)(r % LL); r /= LL;
    int t = (int)(r % TT); r /= TT;
    int b = (int)r;
    float v = 0.0f;
    if (c < 32) {                      // y_ri (B,1,16,T,W,2) -> (nant,ri)
        int nant = c >> 1, ri = c & 1;
        v = y[((((size_t)b * 16 + nant) * TT + t) * (size_t)LL + l) * 2 + ri];
    } else if (c < 288) {              // h_ri (B,1,16,4,2,T,W,2) -> (nant,nue,nspu,ri)
        int cc = c - 32;
        int ri = cc & 1, nspu = (cc >> 1) & 1, nue = (cc >> 2) & 3, nant = cc >> 4;
        v = h[((((((size_t)b * 16 + nant) * 4 + nue) * 2 + nspu) * TT + t) * (size_t)LL + l) * 2 + ri];
    } else if (c < 296) {              // err_var (4,2,T,W)
        int cc = c - 288; int nue = cc >> 1, nspu = cc & 1;
        v = ev[(((size_t)nue * 2 + nspu) * TT + t) * (size_t)LL + l];
    } else if (c == 296) {
        v = no[0];
    }
    X[idx] = (__bf16)v;
}

// ---------------------------------------------------------------------------
// WMMA 3-tap conv-GEMM (bf16 in, f32 accumulate), LDS-staged A panel:
//   Out[s, row, n] = C_init + sum_{tap=0..2} A[s, row+tap-1, :] @ Wt[tap][:, n]
//
// Block (4 waves / 128 thr) stages A rows [row0-1, row0+64] x KP into LDS once
// (boundary zero-fill happens HERE, cooperatively), then the WMMA loop is
// branch-free with EXEC all-ones (a hard WMMA requirement on CDNA5):
//   A-frags <- ds_load_b128 from LDS (3x reuse across taps, no exec masking)
//   B-frags <- global (weights are tiny, L2-resident), 32B contiguous per lane
// Each wave: 4 M-tiles x 2 N-tiles = 64x32 f32 out, 8 v8f accumulators.
//
// Fragment layouts used here (16-bit, wave32; A and B share one contiguous K order):
//   A 16x32 : lane m=l&15, g=l>>4; lo 8 halfwords = K[16g..16g+7], hi = K[16g+8..16g+15]
//   B 32x16 : lane col n=l&15, 16 contiguous K starting at 16*(l>>4)
//   C 16x16 : VGPR r -> row r+8*(l>>4), col l&15
// ---------------------------------------------------------------------------
#define GM 4   // M tiles per wave
#define GN 2   // N tiles per wave
#define GW 4   // waves per block

template<int KP>
__global__ __launch_bounds__(32 * GW)
void convlstm_gemm(const __bf16* __restrict__ A, long long Aseg,
                   const __bf16* __restrict__ Wt, int N,
                   const float* __restrict__ bias,
                   const float* __restrict__ Cmat, long long Cseg,
                   float* __restrict__ Out, long long Oseg) {
    constexpr int ROWS = GM * 16 + 2;        // 64 output rows + conv halo
    constexpr int LDSK = KP + 8;             // pad 16B to stagger LDS banks
    __shared__ __bf16 As[ROWS * LDSK];

    const int tid  = threadIdx.x;
    const int lane = tid & 31;
    const int wave = tid >> 5;
    const int s    = blockIdx.z;
    const int row0 = blockIdx.x * (GM * 16);
    const int n0   = blockIdx.y * (GW * GN * 16) + wave * (GN * 16);
    const int ml   = lane & 15;
    const int g    = lane >> 4;

    // ---- cooperative LDS staging of the A panel (zero-padded halo) ----
    {
        const __bf16* Ap = A + (size_t)s * Aseg;
        constexpr int CPR = KP / 8;                  // 16B chunks per row
        constexpr int TOT = ROWS * CPR;
        for (int ci = tid; ci < TOT; ci += 32 * GW) {
            const int rr = ci / CPR;
            const int kk = (ci % CPR) * 8;
            const int l  = row0 - 1 + rr;            // global row incl. halo
            v8bf v = {};
            if (l >= 0 && l < LL) v = *(const v8bf*)(Ap + (size_t)l * KP + kk);
            *(v8bf*)&As[rr * LDSK + kk] = v;
        }
    }
    __syncthreads();

    // ---- accumulator init: bias and/or precomputed xz slab ----
    v8f acc[GM][GN];
    #pragma unroll
    for (int mt = 0; mt < GM; ++mt)
        #pragma unroll
        for (int nt = 0; nt < GN; ++nt) {
            const int col = n0 + nt * 16 + ml;
            #pragma unroll
            for (int r = 0; r < 8; ++r) {
                float v = bias ? bias[col] : 0.0f;
                if (Cmat) {
                    const int row = row0 + mt * 16 + r + 8 * g;
                    v += Cmat[(size_t)s * Cseg + (size_t)row * N + col];
                }
                acc[mt][nt][r] = v;
            }
        }

    // ---- branch-free WMMA main loop ----
    #pragma unroll
    for (int tap = 0; tap < 3; ++tap) {
        const __bf16* Wtap = Wt + (size_t)tap * N * KP;
        #pragma unroll
        for (int k0 = 0; k0 < KP; k0 += 32) {
            v16bf bf[GN];
            #pragma unroll
            for (int nt = 0; nt < GN; ++nt) {
                const __bf16* bp = Wtap + (size_t)(n0 + nt * 16 + ml) * KP + k0 + g * 16;
                bf[nt] = *(const v16bf*)bp;          // 32B contiguous per lane
            }
            #pragma unroll
            for (int mt = 0; mt < GM; ++mt) {
                const int rr = mt * 16 + ml + tap;   // local row (halo offset -1)
                const __bf16* ap = &As[rr * LDSK + k0];
                union { v16bf v; v8bf h[2]; } u;
                u.h[0] = *(const v8bf*)(ap + 16 * g);       // K[16g .. 16g+7]
                u.h[1] = *(const v8bf*)(ap + 16 * g + 8);   // K[16g+8 .. 16g+15]
                #pragma unroll
                for (int nt = 0; nt < GN; ++nt)
                    acc[mt][nt] = __builtin_amdgcn_wmma_f32_16x16x32_bf16(
                        false, u.v, false, bf[nt], (short)0, acc[mt][nt], false, false);
            }
        }
    }

    // ---- store ----
    #pragma unroll
    for (int mt = 0; mt < GM; ++mt)
        #pragma unroll
        for (int nt = 0; nt < GN; ++nt) {
            const int col = n0 + nt * 16 + ml;
            #pragma unroll
            for (int r = 0; r < 8; ++r) {
                const int row = row0 + mt * 16 + r + 8 * g;
                Out[(size_t)s * Oseg + (size_t)row * N + col] = acc[mt][nt][r];
            }
        }
}

// ---------------------------------------------------------------------------
// LSTM gate math (Keras order i,f,g,o), c/h update, residual, output writes.
// Z: [B, L, 4F] (xz + hz, bias included). Final layer writes d_out directly
// in the transposed [B, NUE, NSPU, (t*L+l)*4 + bit] layout.
// ---------------------------------------------------------------------------
__global__ void lstm_gate(const float* __restrict__ Z, float* __restrict__ C,
                          __bf16* __restrict__ Hst, const __bf16* __restrict__ resid,
                          __bf16* __restrict__ act, float* __restrict__ finout,
                          int F, int t) {
    int idx = blockIdx.x * blockDim.x + threadIdx.x;
    int total = BB * LL * F;
    if (idx >= total) return;
    int f = idx % F;
    int l = (idx / F) % LL;
    int b = idx / (F * LL);

    size_t zb = ((size_t)b * LL + l) * (size_t)(4 * F);
    float zi = Z[zb + f], zf = Z[zb + F + f], zg = Z[zb + 2 * F + f], zo = Z[zb + 3 * F + f];
    auto hsig = [](float x) { return fminf(fmaxf(0.2f * x + 0.5f, 0.0f), 1.0f); };
    float gi = hsig(zi), gf = hsig(zf), gg = tanhf(zg), go = hsig(zo);

    size_t ci = ((size_t)b * LL + l) * F + f;
    float c = gf * C[ci] + gi * gg;
    float h = go * tanhf(c);
    C[ci]   = c;
    Hst[ci] = (__bf16)h;                     // recurrent state for next step's GEMM

    size_t ai = (((size_t)b * TT + t) * LL + l) * F + f;
    float a = h;
    if (resid) a += (float)resid[ai];
    if (act)   act[ai] = (__bf16)a;
    if (finout) {                            // F==32: f = (nue*2+nspu)*4 + bit
        int bit = f & 3, nspu = (f >> 2) & 1, nue = f >> 3;
        size_t oi = (((size_t)b * 4 + nue) * 2 + nspu) * ((size_t)TT * LL * 4)
                  + ((size_t)t * LL + l) * 4 + bit;
        finout[oi] = a;
    }
}

__global__ void zero_u32(unsigned int* __restrict__ p, long long n) {
    long long i = (long long)blockIdx.x * blockDim.x + threadIdx.x;
    if (i < n) p[i] = 0u;
}

// ---------------------------------------------------------------------------
// Host driver
// ---------------------------------------------------------------------------
extern "C" void kernel_launch(void* const* d_in, const int* in_sizes, int n_in,
                              void* d_out, int out_size, void* d_ws, size_t ws_size,
                              hipStream_t stream) {
    const float* y_ri  = (const float*)d_in[0];
    const float* h_ri  = (const float*)d_in[1];
    const float* evar  = (const float*)d_in[2];
    const float* no    = (const float*)d_in[3];
    const float* Wx_in = (const float*)d_in[4];
    const float* Wh_in = (const float*)d_in[5];
    const float* b_in  = (const float*)d_in[6];
    const float* Wx_r0 = (const float*)d_in[7];
    const float* Wh_r0 = (const float*)d_in[8];
    const float* b_r0  = (const float*)d_in[9];
    const float* Wx_r1 = (const float*)d_in[10];
    const float* Wh_r1 = (const float*)d_in[11];
    const float* b_r1  = (const float*)d_in[12];
    const float* Wx_o  = (const float*)d_in[13];
    const float* Wh_o  = (const float*)d_in[14];
    const float* b_o   = (const float*)d_in[15];

    char* ws = (char*)d_ws;
    size_t off = 0;
    auto alloc = [&](size_t bytes) -> void* {
        off = (off + 255) & ~(size_t)255;
        void* p = ws + off;
        off += bytes;
        return p;
    };

    __bf16* X0  = (__bf16*)alloc((size_t)BB * TT * LL * C0P * 2);
    float*  XZ  = (float*) alloc((size_t)BB * TT * LL * N1 * 4);   // reused per layer
    float*  ZT  = (float*) alloc((size_t)BB * LL * N1 * 4);        // per-timestep gate input
    __bf16* Z1  = (__bf16*)alloc((size_t)BB * TT * LL * F1 * 2);
    __bf16* Z2  = (__bf16*)alloc((size_t)BB * TT * LL * F1 * 2);
    __bf16* Z3  = (__bf16*)alloc((size_t)BB * TT * LL * F1 * 2);
    __bf16* H   = (__bf16*)alloc((size_t)BB * LL * F1 * 2);
    float*  Cst = (float*) alloc((size_t)BB * LL * F1 * 4);
    __bf16* WtX1 = (__bf16*)alloc((size_t)3 * N1 * C0P * 2);
    __bf16* WtH1 = (__bf16*)alloc((size_t)3 * N1 * F1 * 2);
    __bf16* WtX2 = (__bf16*)alloc((size_t)3 * N1 * F1 * 2);
    __bf16* WtH2 = (__bf16*)alloc((size_t)3 * N1 * F1 * 2);
    __bf16* WtX3 = (__bf16*)alloc((size_t)3 * N1 * F1 * 2);
    __bf16* WtH3 = (__bf16*)alloc((size_t)3 * N1 * F1 * 2);
    __bf16* WtX4 = (__bf16*)alloc((size_t)3 * N4 * F1 * 2);
    __bf16* WtH4 = (__bf16*)alloc((size_t)3 * N4 * F4 * 2);
    (void)ws_size; (void)in_sizes; (void)n_in; (void)out_size;

    auto repack = [&](const float* src, __bf16* dst, int Cin, int Kp, int Nout) {
        long long n = 3LL * Nout * Kp;
        repack_w<<<dim3((unsigned)((n + 255) / 256)), 256, 0, stream>>>(src, dst, Cin, Kp, Nout);
    };
    repack(Wx_in, WtX1, C0, C0P, N1);
    repack(Wh_in, WtH1, F1, F1, N1);
    repack(Wx_r0, WtX2, F1, F1, N1);
    repack(Wh_r0, WtH2, F1, F1, N1);
    repack(Wx_r1, WtX3, F1, F1, N1);
    repack(Wh_r1, WtH3, F1, F1, N1);
    repack(Wx_o,  WtX4, F1, F1, N4);
    repack(Wh_o,  WtH4, F4, F4, N4);

    {
        long long n = (long long)BB * TT * LL * C0P;
        pack_input<<<dim3((unsigned)((n + 255) / 256)), 256, 0, stream>>>(y_ri, h_ri, evar, no, X0);
    }

    // KP-templated GEMM dispatch (KP in {320, 64, 32})
    auto launch_gemm = [&](int KP, dim3 grid, const __bf16* A, long long Aseg,
                           const __bf16* Wt, int N, const float* bias,
                           const float* Cmat, long long Cseg,
                           float* Out, long long Oseg) {
        if (KP == C0P)
            convlstm_gemm<C0P><<<grid, 32 * GW, 0, stream>>>(A, Aseg, Wt, N, bias, Cmat, Cseg, Out, Oseg);
        else if (KP == F1)
            convlstm_gemm<F1><<<grid, 32 * GW, 0, stream>>>(A, Aseg, Wt, N, bias, Cmat, Cseg, Out, Oseg);
        else
            convlstm_gemm<F4><<<grid, 32 * GW, 0, stream>>>(A, Aseg, Wt, N, bias, Cmat, Cseg, Out, Oseg);
    };

    auto run_layer = [&](const __bf16* Xsrc, int Kin, const __bf16* WtX, const __bf16* WtH,
                         const float* bias, int F, int N,
                         const __bf16* resid, __bf16* actout, float* finout) {
        // input-to-hidden conv for all (b,t) at once, bias folded into C-init
        dim3 gx(LL / (GM * 16), N / (GW * GN * 16), BB * TT);
        launch_gemm(Kin, gx, Xsrc, (long long)LL * Kin, WtX, N, bias,
                    nullptr, 0, XZ, (long long)LL * N);
        // zero recurrent state (deterministic each call)
        long long hn = (long long)BB * LL * F / 2;   // bf16 as u32
        long long cn = (long long)BB * LL * F;
        zero_u32<<<dim3((unsigned)((hn + 255) / 256)), 256, 0, stream>>>((unsigned int*)H, hn);
        zero_u32<<<dim3((unsigned)((cn + 255) / 256)), 256, 0, stream>>>((unsigned int*)Cst, cn);
        // sequential scan over T: hidden conv GEMM (C-init = xz slab) + gates
        for (int t = 0; t < TT; ++t) {
            dim3 gh(LL / (GM * 16), N / (GW * GN * 16), BB);
            launch_gemm(F, gh, H, (long long)LL * F, WtH, N, nullptr,
                        XZ + (size_t)t * LL * N, (long long)TT * LL * N,
                        ZT, (long long)LL * N);
            int gn = BB * LL * F;
            lstm_gate<<<dim3((gn + 255) / 256), 256, 0, stream>>>(
                ZT, Cst, H, resid, actout, finout, F, t);
        }
    };

    run_layer(X0, C0P, WtX1, WtH1, b_in, F1, N1, nullptr, Z1, nullptr);
    run_layer(Z1, F1,  WtX2, WtH2, b_r0, F1, N1, Z1,      Z2, nullptr);
    run_layer(Z2, F1,  WtX3, WtH3, b_r1, F1, N1, Z2,      Z3, nullptr);
    run_layer(Z3, F1,  WtX4, WtH4, b_o,  F4, N4, nullptr, nullptr, (float*)d_out);
}
